// Attention_14620068676191
// MI455X (gfx1250) — matrix operand order repaired and checked by reading, not compiled, not run on hardware
//
#include <hip/hip_runtime.h>
#include <hip/hip_bf16.h>

// Problem constants (match the JAX reference)
#define SEQ   2048
#define BATCH 8
#define DIM   1024

typedef __attribute__((ext_vector_type(16))) __bf16 v16bf;
typedef __attribute__((ext_vector_type(8)))  float  v8f;
typedef __attribute__((ext_vector_type(4)))  __bf16 v4bf;
typedef int v4i __attribute__((vector_size(16)));   // matches builtin's pointee type

union FragBF { v16bf v; uint4 u[2]; };

#define BM 128
#define BN 128
#define BK 32
#define LDSK 40   // padded LDS row stride in bf16 elements (80 B, 16B-aligned)

#define AS1 __attribute__((address_space(1)))
#define AS3 __attribute__((address_space(3)))

#if defined(__gfx1250__) && __has_builtin(__builtin_amdgcn_global_load_async_to_lds_b128)
#define ASYNC_LDS 1
#else
#define ASYNC_LDS 0
#endif

// ---------------------------------------------------------------- converts
__global__ __launch_bounds__(256) void k_f32_to_bf16(const float* __restrict__ in,
                                                     __bf16* __restrict__ out, int n) {
  int i = (blockIdx.x * blockDim.x + threadIdx.x) * 4;
  if (i < n) {
    float4 f = *(const float4*)(in + i);
    v4bf h;
    h.x = (__bf16)f.x; h.y = (__bf16)f.y; h.z = (__bf16)f.z; h.w = (__bf16)f.w;
    *(v4bf*)(out + i) = h;
  }
}

__global__ __launch_bounds__(256) void k_zero_f32(float* __restrict__ p, int n) {
  int i = blockIdx.x * blockDim.x + threadIdx.x;
  if (i < n) p[i] = 0.f;
}

// ------------------------------------------------------- async LDS staging
// Copy one 64-byte tile row (32 bf16) global -> LDS.
__device__ __forceinline__ void copy_row64(const __bf16* g, __bf16* l) {
#if ASYNC_LDS
#pragma unroll
  for (int c = 0; c < 4; ++c)
    __builtin_amdgcn_global_load_async_to_lds_b128(
        (AS1 v4i*)(void*)(g + c * 8), (AS3 v4i*)(void*)(l + c * 8), 0, 0);
#else
  uint4* d = (uint4*)l;
  const uint4* s = (const uint4*)g;
#pragma unroll
  for (int c = 0; c < 4; ++c) d[c] = s[c];
#endif
}

__device__ __forceinline__ void wait_async0() {
#if ASYNC_LDS
#if __has_builtin(__builtin_amdgcn_s_wait_asynccnt)
  __builtin_amdgcn_s_wait_asynccnt(0);
#else
  asm volatile("s_wait_asynccnt 0x0" ::: "memory");
#endif
#endif
}

// ---------------------------------------------------------------- fragments
// Two ds_load_b128 per fragment per the CDNA5 16-bit WMMA VGPR layouts.
__device__ __forceinline__ v16bf frag_from_lds(const __bf16* rowp, int off1, int off2) {
  FragBF f;
  f.u[0] = *(const uint4*)(rowp + off1);
  f.u[1] = *(const uint4*)(rowp + off2);
  return f.v;
}

// ---------------------------------------------------------------- GEMM
// 128 threads = 4 waves, waves 2x2; each wave owns a 64x64 sub-tile =
// 4x4 WMMA tiles (128 acc VGPRs), 16 ds_load_b128 : 16 v_wmma per k-step.
// Double-buffered LDS, async global->LDS prefetch of k+1 overlaps compute.
//
// EPI 0: projection  C[m,n] -> q[(b*SEQ+s)*DIM+n] bf16, m = s*BATCH+b
// EPI 1: scores      p = exp(tanh(C/32)) -> P[b][m][n] bf16, rowsum[b][m] += p
// EPI 2: PV          out[(m*BATCH+b)*DIM+n] = C / rowsum[b][m]  (f32)
template <int EPI>
__global__ __launch_bounds__(128) void k_gemm(const __bf16* __restrict__ A, int lda,
                                              const __bf16* __restrict__ Bsrc, int ldb,
                                              int Kdim,
                                              __bf16* __restrict__ outBF,
                                              float* __restrict__ outF,
                                              float* __restrict__ rowsum) {
  __shared__ __bf16 sA[2][BM * LDSK];
  __shared__ __bf16 sB[2][BN * LDSK];
  __shared__ float  sSum[BM];

  const int tid  = threadIdx.x;
  const int lane = tid & 31;
  const int wave = tid >> 5;     // 0..3
  const int wm   = wave >> 1;    // 0..1 -> 4 m-tiles each (64 rows)
  const int wn   = wave & 1;     // 0..1 -> 4 n-tiles each (64 cols)
  const int bm0  = blockIdx.x * BM;
  const int bn0  = blockIdx.y * BN;
  const int b    = blockIdx.z;

  const __bf16* Aeff = A;
  const __bf16* Beff = Bsrc;
  if constexpr (EPI == 1) {
    Aeff = A + (size_t)b * SEQ * DIM;
    Beff = Bsrc + (size_t)b * SEQ * DIM;
  } else if constexpr (EPI == 2) {
    Aeff = A + (size_t)b * SEQ * SEQ;   // P for this batch
    Beff = Bsrc + (size_t)b * DIM;      // xb column base for this batch
  }

  if constexpr (EPI == 1) {
    if (tid < BM) sSum[tid] = 0.f;
  }

  // stage one 32-K slice into LDS buffer `buf`
  auto stage = [&](int k0, int buf) {
    copy_row64(Aeff + (size_t)(bm0 + tid) * lda + k0, &sA[buf][tid * LDSK]);
    if constexpr (EPI != 2) {
      copy_row64(Beff + (size_t)(bn0 + tid) * ldb + k0, &sB[buf][tid * LDSK]);
    } else {
      // B[n=d][k=t] = xb[t,b,d]: 64B row chunk of x, transpose-scatter to LDS
      const int t  = tid & 31;          // k within tile
      const int nc = (tid >> 5) * 32;   // n chunk base
      const __bf16* g = Beff + (size_t)(k0 + t) * (BATCH * DIM) + bn0 + nc;
      uint4 v[4];
#pragma unroll
      for (int c = 0; c < 4; ++c) v[c] = ((const uint4*)g)[c];
      const __bf16* e = (const __bf16*)v;
#pragma unroll
      for (int j = 0; j < 32; ++j) sB[buf][(nc + j) * LDSK + t] = e[j];
    }
  };

  const v8f zacc = {0.f, 0.f, 0.f, 0.f, 0.f, 0.f, 0.f, 0.f};
  v8f acc[4][4];
#pragma unroll
  for (int i = 0; i < 4; ++i)
#pragma unroll
    for (int j = 0; j < 4; ++j) acc[i][j] = zacc;

  const int mrow  = lane & 15;
  const int aoff1 = (lane & 16) ? 8 : 0;    // A: chunks at {h, h+16}
  const int boff1 = (lane & 16) ? 8 : 0;    // B: chunks at {h, h+16}

  const int nk = Kdim / BK;
  stage(0, 0);
  for (int kt = 0; kt < nk; ++kt) {
    const int cur = kt & 1;
    wait_async0();       // our async loads for buf[cur] have landed
    __syncthreads();     // everyone's have; prior reads of buf[1-cur] done
    if (kt + 1 < nk) stage((kt + 1) * BK, 1 - cur);  // overlaps compute below

    v16bf af[4];
#pragma unroll
    for (int i = 0; i < 4; ++i)
      af[i] = frag_from_lds(&sA[cur][(wm * 64 + i * 16 + mrow) * LDSK], aoff1, aoff1 + 16);
#pragma unroll
    for (int j = 0; j < 4; ++j) {
      const v16bf bfm =
          frag_from_lds(&sB[cur][(wn * 64 + j * 16 + mrow) * LDSK], boff1, boff1 + 16);
#pragma unroll
      for (int i = 0; i < 4; ++i)
        acc[i][j] = __builtin_amdgcn_wmma_f32_16x16x32_bf16(
            false, af[i], false, bfm, (short)0, acc[i][j], false, false);
    }
  }

  // ------------------------------------------------------------- epilogue
  const int mbase = (lane & 16) ? 8 : 0;
  const int nlane = lane & 15;
#pragma unroll
  for (int i = 0; i < 4; ++i) {
#pragma unroll
    for (int r = 0; r < 8; ++r) {
      const int m = bm0 + wm * 64 + i * 16 + mbase + r;
      if constexpr (EPI == 0) {
        const int s = m >> 3, bb = m & 7;  // m = s*BATCH + b
        __bf16* rowp = outBF + ((size_t)bb * SEQ + s) * DIM;
#pragma unroll
        for (int j = 0; j < 4; ++j)
          rowp[bn0 + wn * 64 + j * 16 + nlane] = (__bf16)acc[i][j][r];
      } else if constexpr (EPI == 1) {
        __bf16* rowp = outBF + (size_t)b * SEQ * SEQ + (size_t)m * SEQ;
        float s = 0.f;
#pragma unroll
        for (int j = 0; j < 4; ++j) {
          const float p = __expf(tanhf(acc[i][j][r] * (1.f / 32.f)));  // scale=sqrt(1024)
          rowp[bn0 + wn * 64 + j * 16 + nlane] = (__bf16)p;
          s += p;
        }
        atomicAdd(&sSum[m - bm0], s);
      } else {
        const float inv = 1.f / rowsum[b * SEQ + m];
        float* rowp = outF + ((size_t)m * BATCH + b) * DIM;
#pragma unroll
        for (int j = 0; j < 4; ++j)
          rowp[bn0 + wn * 64 + j * 16 + nlane] = acc[i][j][r] * inv;
      }
    }
  }
  if constexpr (EPI == 1) {
    __syncthreads();
    if (tid < BM) atomicAdd(&rowsum[b * SEQ + bm0 + tid], sSum[tid]);
  }
}

// ---------------------------------------------------------------- launcher
extern "C" void kernel_launch(void* const* d_in, const int* in_sizes, int n_in,
                              void* d_out, int out_size, void* d_ws, size_t ws_size,
                              hipStream_t stream) {
  const float* x  = (const float*)d_in[0];
  const float* Wq = (const float*)d_in[1];
  const float* Wk = (const float*)d_in[2];
  float* out = (float*)d_out;

  char* ws = (char*)d_ws;
  size_t off = 0;
  __bf16* xb  = (__bf16*)(ws + off); off += (size_t)SEQ * BATCH * DIM * 2;   // 32 MB
  __bf16* wqb = (__bf16*)(ws + off); off += (size_t)DIM * DIM * 2;           //  2 MB
  __bf16* wkb = (__bf16*)(ws + off); off += (size_t)DIM * DIM * 2;           //  2 MB
  __bf16* qb  = (__bf16*)(ws + off); off += (size_t)BATCH * SEQ * DIM * 2;   // 32 MB
  __bf16* kb  = (__bf16*)(ws + off); off += (size_t)BATCH * SEQ * DIM * 2;   // 32 MB
  __bf16* P   = (__bf16*)(ws + off); off += (size_t)BATCH * SEQ * SEQ * 2;   // 64 MB
  float*  rsum = (float*)(ws + off); off += (size_t)BATCH * SEQ * 4;         // 64 KB

  const int nX = SEQ * BATCH * DIM;   // 16M
  const int nW = DIM * DIM;           // 1M
  k_f32_to_bf16<<<nX / 1024, 256, 0, stream>>>(x, xb, nX);
  k_f32_to_bf16<<<nW / 1024, 256, 0, stream>>>(Wq, wqb, nW);
  k_f32_to_bf16<<<nW / 1024, 256, 0, stream>>>(Wk, wkb, nW);
  k_zero_f32<<<(BATCH * SEQ) / 256, 256, 0, stream>>>(rsum, BATCH * SEQ);

  // q,k projections: M = SEQ*BATCH flattened rows of x, N = K = DIM
  dim3 g1((SEQ * BATCH) / BM, DIM / BN, 1);
  k_gemm<0><<<g1, 128, 0, stream>>>(xb, DIM, wqb, DIM, DIM, qb, nullptr, nullptr);
  k_gemm<0><<<g1, 128, 0, stream>>>(xb, DIM, wkb, DIM, DIM, kb, nullptr, nullptr);

  // scores + tanh + exp + rowsum, per batch
  dim3 g2(SEQ / BM, SEQ / BN, BATCH);
  k_gemm<1><<<g2, 128, 0, stream>>>(qb, DIM, kb, DIM, DIM, P, nullptr, rsum);

  // out = (P @ x_b) / rowsum, per batch
  dim3 g3(SEQ / BM, DIM / BN, BATCH);
  k_gemm<2><<<g3, 128, 0, stream>>>(P, SEQ, xb, 0, SEQ, nullptr, out, rsum);

  (void)in_sizes; (void)n_in; (void)out_size; (void)ws_size;
}
